// GraphConv_25632364822910
// MI455X (gfx1250) — compile-verified, matrix operand order checked by reading, not run on hardware
//
#include <hip/hip_runtime.h>

#define IC 256
#define FF 256
#define EC 2

typedef __attribute__((ext_vector_type(16))) __bf16 v16bf;
typedef __attribute__((ext_vector_type(8)))  float  v8f;

union AFrag { v16bf v; __bf16 e[16]; };
union BFrag { v16bf v; uint4  q[2]; };

// ---------------------------------------------------------------------------
// Prep: Wt[n][k] = bf16(W[k][n])   (128 KB, L2/WGP$-resident for the GEMM)
// ---------------------------------------------------------------------------
__global__ void wt_prep_kernel(const float* __restrict__ W,
                               __bf16* __restrict__ Wt) {
    int idx = blockIdx.x * blockDim.x + threadIdx.x;   // IC*FF threads
    int n = idx >> 8;          // output column (F)
    int k = idx & 255;         // input channel
    Wt[(size_t)n * IC + k] = (__bf16)W[(size_t)k * FF + n];
}

// ---------------------------------------------------------------------------
// GEMM: xdw[N,256] = bf16(x) @ bf16(W) + b_dense, f32 accumulate via WMMA.
// One wave per 16-row strip; wave holds all 16 N-tile accumulators (128 VGPRs)
// so X is streamed once. B fragments are double-buffered so Wt loads overlap
// with WMMA issue instead of forcing s_wait_loadcnt 0 before every matmul.
// ---------------------------------------------------------------------------
__global__ void __launch_bounds__(256)
gemm_xw_kernel(const float* __restrict__ x, const __bf16* __restrict__ Wt,
               const float* __restrict__ bdense, float* __restrict__ xdw,
               int nRowTiles, int N) {
    const int wid  = (blockIdx.x * blockDim.x + threadIdx.x) >> 5;
    const int lane = threadIdx.x & 31;
    if (wid >= nRowTiles) return;

    const int half = lane >> 4;      // K-group selector (A/B), M-group (C/D)
    const int m    = lane & 15;      // A: row within tile; B/C/D: column N

    int rowA = wid * 16 + m;
    if (rowA >= N) rowA = N - 1;                 // tail guard (N=100000 is exact)
    const float* __restrict__ xrow = x + (size_t)rowA * IC;

    v8f acc[16] = {};                            // 16 N-tiles of 16x16 f32 C/D

    #pragma unroll
    for (int kb = 0; kb < IC; kb += 32) {
        const int k0 = kb + half * 8;
        // A fragment: lane holds x[rowA, k0..k0+7] and x[rowA, k0+16..k0+23]
        float4 f0 = *(const float4*)(xrow + k0);
        float4 f1 = *(const float4*)(xrow + k0 + 4);
        float4 f2 = *(const float4*)(xrow + k0 + 16);
        float4 f3 = *(const float4*)(xrow + k0 + 20);
        AFrag a;
        a.e[0]=(__bf16)f0.x;  a.e[1]=(__bf16)f0.y;  a.e[2]=(__bf16)f0.z;  a.e[3]=(__bf16)f0.w;
        a.e[4]=(__bf16)f1.x;  a.e[5]=(__bf16)f1.y;  a.e[6]=(__bf16)f1.z;  a.e[7]=(__bf16)f1.w;
        a.e[8]=(__bf16)f2.x;  a.e[9]=(__bf16)f2.y;  a.e[10]=(__bf16)f2.z; a.e[11]=(__bf16)f2.w;
        a.e[12]=(__bf16)f3.x; a.e[13]=(__bf16)f3.y; a.e[14]=(__bf16)f3.z; a.e[15]=(__bf16)f3.w;

        // B fragments: lane's column N = m, same K packing as A, from Wt.
        const __bf16* __restrict__ wbase = Wt + (size_t)m * IC + k0;

        BFrag b[2];
        b[0].q[0] = *(const uint4*)(wbase);        // nt=0: K = k0 .. k0+7
        b[0].q[1] = *(const uint4*)(wbase + 16);   //       K = k0+16 .. k0+23

        #pragma unroll
        for (int nt = 0; nt < 16; ++nt) {
            const int cur = nt & 1;
            if (nt < 15) {                         // prefetch next B tile
                const __bf16* __restrict__ wn =
                    wbase + (size_t)(nt + 1) * (16 * IC);
                b[cur ^ 1].q[0] = *(const uint4*)(wn);
                b[cur ^ 1].q[1] = *(const uint4*)(wn + 16);
            }
            acc[nt] = __builtin_amdgcn_wmma_f32_16x16x32_bf16(
                /*neg_a=*/false, a.v, /*neg_b=*/false, b[cur].v,
                /*c_mod=*/(short)0, acc[nt],
                /*reuse_a=*/false, /*reuse_b=*/false);
        }
    }

    // Store D + b_dense. C/D layout: VGPR r -> M = 8*half + r, N = m.
    #pragma unroll
    for (int nt = 0; nt < 16; ++nt) {
        const int   col = nt * 16 + m;
        const float bv  = bdense[col];
        #pragma unroll
        for (int r = 0; r < 8; ++r) {
            const int rowOut = wid * 16 + 8 * half + r;
            if (rowOut < N)
                xdw[(size_t)rowOut * FF + col] = acc[nt][r] + bv;
        }
    }
}

// ---------------------------------------------------------------------------
// Initialize out[n, 512] = bias[c]  (float4-vectorized broadcast)
// ---------------------------------------------------------------------------
__global__ void bias_init_kernel(float* __restrict__ out,
                                 const float* __restrict__ bias, int n4) {
    int idx = blockIdx.x * blockDim.x + threadIdx.x;  // N*128 float4 elements
    if (idx >= n4) return;
    ((float4*)out)[idx] = ((const float4*)bias)[idx & 127];
}

// ---------------------------------------------------------------------------
// Edge scatter: one wave per edge. Lane l owns channels [8l, 8l+8).
// Edge index forced uniform via readfirstlane so src/dst/w become scalar
// (s_load) reads; gather is two 128-bit vector loads from L2-resident xdw;
// scatter is 8 native global_atomic_add_f32 per lane.
// ---------------------------------------------------------------------------
__global__ void __launch_bounds__(256)
edge_scatter_kernel(const int* __restrict__ esrc, const int* __restrict__ edst,
                    const float* __restrict__ ew, const float* __restrict__ xdw,
                    float* __restrict__ out, int ePer, int totalEdges) {
    const int wid  = blockIdx.x * 8 + (threadIdx.x >> 5);  // wave-uniform
    const int lane = threadIdx.x & 31;
    if (wid >= totalEdges) return;

    const int e = __builtin_amdgcn_readfirstlane(wid);     // force SGPR
    const int t = (e >= ePer) ? 1 : 0;                     // edge type (EC==2)

    const int   src = esrc[e];                             // scalar loads
    const int   dst = edst[e];
    const float w   = ew[e];

    const float4* __restrict__ row =
        (const float4*)(xdw + (size_t)src * FF) + lane * 2;
    const float4 a = row[0];
    const float4 b = row[1];

    float* __restrict__ o = out + (size_t)dst * (EC * FF) + t * FF + lane * 8;
    unsafeAtomicAdd(o + 0, w * a.x);
    unsafeAtomicAdd(o + 1, w * a.y);
    unsafeAtomicAdd(o + 2, w * a.z);
    unsafeAtomicAdd(o + 3, w * a.w);
    unsafeAtomicAdd(o + 4, w * b.x);
    unsafeAtomicAdd(o + 5, w * b.y);
    unsafeAtomicAdd(o + 6, w * b.z);
    unsafeAtomicAdd(o + 7, w * b.w);
}

// ---------------------------------------------------------------------------
// ReLU sweep (float4-vectorized)
// ---------------------------------------------------------------------------
__global__ void relu_kernel(float* __restrict__ out, int n4) {
    int idx = blockIdx.x * blockDim.x + threadIdx.x;
    if (idx >= n4) return;
    float4 v = ((float4*)out)[idx];
    v.x = fmaxf(v.x, 0.0f); v.y = fmaxf(v.y, 0.0f);
    v.z = fmaxf(v.z, 0.0f); v.w = fmaxf(v.w, 0.0f);
    ((float4*)out)[idx] = v;
}

// ---------------------------------------------------------------------------
extern "C" void kernel_launch(void* const* d_in, const int* in_sizes, int n_in,
                              void* d_out, int out_size, void* d_ws, size_t ws_size,
                              hipStream_t stream) {
    const float* x      = (const float*)d_in[0];
    const int*   esrc   = (const int*)  d_in[1];
    const int*   edst   = (const int*)  d_in[2];
    const float* ew     = (const float*)d_in[3];
    const float* W      = (const float*)d_in[4];
    const float* bdense = (const float*)d_in[5];
    const float* bias   = (const float*)d_in[6];
    float*       out    = (float*)d_out;

    const int N    = in_sizes[0] / IC;       // 100000
    const int ePer = in_sizes[1] / EC;       // 3200000

    // Scratch layout: [0,128KB) Wt bf16; then XDW f32 [N,256]
    __bf16* Wt  = (__bf16*)d_ws;
    float*  xdw = (float*)((char*)d_ws + (size_t)IC * FF * sizeof(__bf16));

    // 1) Transpose+convert W -> bf16 Wt (tiny)
    wt_prep_kernel<<<(IC * FF) / 256, 256, 0, stream>>>(W, Wt);

    // 2) WMMA GEMM: one wave per 16-row strip
    const int nRowTiles = (N + 15) / 16;
    const int gemmBlocks = (nRowTiles * 32 + 255) / 256;
    gemm_xw_kernel<<<gemmBlocks, 256, 0, stream>>>(x, Wt, bdense, xdw,
                                                   nRowTiles, N);

    // 3) out = bias (broadcast init so scatter accumulates on top)
    const int n4 = N * (EC * FF) / 4;        // N*128 float4s
    bias_init_kernel<<<(n4 + 255) / 256, 256, 0, stream>>>(out, bias, n4);

    // 4) Edge gather/scale/atomic-scatter, one wave per edge, both edge types
    const int totalEdges = EC * ePer;
    const int scatterBlocks = (totalEdges + 7) / 8;   // 8 waves (edges) / block
    edge_scatter_kernel<<<scatterBlocks, 256, 0, stream>>>(
        esrc, edst, ew, xdw, out, ePer, totalEdges);

    // 5) ReLU
    relu_kernel<<<(n4 + 255) / 256, 256, 0, stream>>>(out, n4);
}
